// MultiHeadAttention_8169027797124
// MI455X (gfx1250) — matrix operand order repaired and checked by reading, not compiled, not run on hardware
//
#include <hip/hip_runtime.h>
#include <math.h>

// ---- problem constants (match reference) ----
#define SEQ   4096
#define EMB   512
#define NH    8
#define HD    64

typedef __attribute__((ext_vector_type(16))) __bf16          v16bf;
typedef __attribute__((ext_vector_type(8)))  float           v8f;
typedef __attribute__((ext_vector_type(4)))  float           v4f;
typedef __attribute__((ext_vector_type(16))) unsigned short  v16u;
typedef __attribute__((ext_vector_type(8)))  unsigned short  v8u;

// float -> bf16 bits, round-to-nearest-even
__device__ __forceinline__ unsigned short f2bf(float f) {
  unsigned u = __float_as_uint(f);
  unsigned r = u + 0x7FFFu + ((u >> 16) & 1u);
  return (unsigned short)(r >> 16);
}

// ---- WMMA fragment loaders (CDNA5 16x16x32 bf16 layouts, CDNA5 ISA) ----
// A-matrix 16x32 (MxK), row-major source, lda in elements.
// lane L: m = L&15 ; lanes 0-15 K-base 0, lanes 16-31 K-base 8.
// elements 0..7  -> K = kb + koff + 0..7      (VGPR0-3)
// elements 8..15 -> K = kb + 16 + koff + 0..7 (VGPR4-7)
__device__ __forceinline__ v16bf load_a_bf16(const unsigned short* A, int lda,
                                             int kb, int lane) {
  int m    = lane & 15;
  int koff = (lane >> 4) << 3;                 // 0 or 8
  const unsigned short* p = A + m * lda + kb + koff;
  v8u lo = *(const v8u*)(p);
  v8u hi = *(const v8u*)(p + 16);
  v16u t;
#pragma unroll
  for (int i = 0; i < 8; ++i) { t[i] = lo[i]; t[i + 8] = hi[i]; }
  return __builtin_bit_cast(v16bf, t);
}

// Same A fragment but source is fp32 row-major (attention probabilities).
__device__ __forceinline__ v16bf load_a_f32(const float* A, int lda,
                                            int kb, int lane) {
  int m    = lane & 15;
  int koff = (lane >> 4) << 3;
  const float* p = A + m * lda + kb + koff;
  v4f a0 = *(const v4f*)(p);
  v4f a1 = *(const v4f*)(p + 4);
  v4f b0 = *(const v4f*)(p + 16);
  v4f b1 = *(const v4f*)(p + 20);
  v16u t;
#pragma unroll
  for (int i = 0; i < 4; ++i) {
    t[i]      = f2bf(a0[i]);
    t[i + 4]  = f2bf(a1[i]);
    t[i + 8]  = f2bf(b0[i]);
    t[i + 12] = f2bf(b1[i]);
  }
  return __builtin_bit_cast(v16bf, t);
}

// B-matrix 32x16 (KxN) from a transposed (N-major, N x K) source, ldb in elems.
// lane L: n = L&15 ; same K order as A: elements 0..7 -> K = kb+koff+0..7,
// elements 8..15 -> K = kb+16+koff+0..7 (koff = 0 or 8).
__device__ __forceinline__ v16bf load_b_bf16(const unsigned short* BT, int ldb,
                                             int kb, int lane) {
  int n  = lane & 15;
  int koff = (lane >> 4) << 3;                 // 0 or 8
  const unsigned short* p = BT + n * ldb + kb + koff;
  v8u lo = *(const v8u*)(p);
  v8u hi = *(const v8u*)(p + 16);
  v16u t;
#pragma unroll
  for (int i = 0; i < 8; ++i) { t[i] = lo[i]; t[i + 8] = hi[i]; }
  return __builtin_bit_cast(v16bf, t);
}

__device__ __forceinline__ v8f wmma_bf16(v16bf a, v16bf b, v8f c) {
  return __builtin_amdgcn_wmma_f32_16x16x32_bf16(false, a, false, b,
                                                 (short)0, c, false, false);
}

// ============ kernel 0: fp32 -> bf16 conversions + weight transposes ============
__global__ __launch_bounds__(256) void mha_prep_kernel(
    const float* __restrict__ x,
    const float* __restrict__ Wq, const float* __restrict__ Wk,
    const float* __restrict__ Wv, const float* __restrict__ Wo,
    unsigned short* __restrict__ x_bf,
    unsigned short* __restrict__ wqt, unsigned short* __restrict__ wkt,
    unsigned short* __restrict__ wvt, unsigned short* __restrict__ wot) {
  int i = blockIdx.x * blockDim.x + threadIdx.x;
  if (i < SEQ * EMB) x_bf[i] = f2bf(x[i]);
  if (i < EMB * EMB) {
    int n = i / EMB, k = i % EMB;              // wt[n][k] = W[k][n]
    wqt[i] = f2bf(Wq[k * EMB + n]);
    wkt[i] = f2bf(Wk[k * EMB + n]);
    wvt[i] = f2bf(Wv[k * EMB + n]);
    wot[i] = f2bf(Wo[k * EMB + n]);
  }
}

// ============ kernel 1: Q/K/V projections (16x64 tile per wave, SW-pipelined) ===
// p=0 -> Q (H,S,D) ; p=1 -> K (H,S,D) ; p=2 -> V transposed (H,D,S)
__global__ __launch_bounds__(256) void mha_qkv_kernel(
    const unsigned short* __restrict__ x_bf,
    const unsigned short* __restrict__ wqt, const unsigned short* __restrict__ wkt,
    const unsigned short* __restrict__ wvt,
    const float* __restrict__ bq, const float* __restrict__ bk,
    const float* __restrict__ bv,
    unsigned short* __restrict__ q_bf, unsigned short* __restrict__ k_bf,
    unsigned short* __restrict__ vt_bf) {
  int lane = threadIdx.x & 31;
  int wid  = (blockIdx.x * blockDim.x + threadIdx.x) >> 5;
  int p  = wid >> 11;                          // projection: 0..2 (2048 tiles each)
  int t  = wid & 2047;
  int nb = t & 7;                              // 8 n-blocks of 64
  int mb = t >> 3;                             // 256 m-blocks of 16
  const unsigned short* wt   = (p == 0) ? wqt : (p == 1) ? wkt : wvt;
  const float*          bias = (p == 0) ? bq  : (p == 1) ? bk  : bv;

  const unsigned short* A  = x_bf + (size_t)mb * 16 * EMB;
  const unsigned short* B0 = wt + (size_t)(nb * 64) * EMB;

  v8f c[4] = {};
  // prologue: fragments for kb = 0
  v16bf a = load_a_bf16(A, EMB, 0, lane);
  v16bf b0 = load_b_bf16(B0,            EMB, 0, lane);
  v16bf b1 = load_b_bf16(B0 + 16 * EMB, EMB, 0, lane);
  v16bf b2 = load_b_bf16(B0 + 32 * EMB, EMB, 0, lane);
  v16bf b3 = load_b_bf16(B0 + 48 * EMB, EMB, 0, lane);
  for (int kb = 0; kb < EMB; kb += 32) {
    int kn = (kb + 32 < EMB) ? kb + 32 : kb;   // preload next k-step
    v16bf an  = load_a_bf16(A, EMB, kn, lane);
    v16bf bn0 = load_b_bf16(B0,            EMB, kn, lane);
    v16bf bn1 = load_b_bf16(B0 + 16 * EMB, EMB, kn, lane);
    v16bf bn2 = load_b_bf16(B0 + 32 * EMB, EMB, kn, lane);
    v16bf bn3 = load_b_bf16(B0 + 48 * EMB, EMB, kn, lane);
    c[0] = wmma_bf16(a, b0, c[0]);
    c[1] = wmma_bf16(a, b1, c[1]);
    c[2] = wmma_bf16(a, b2, c[2]);
    c[3] = wmma_bf16(a, b3, c[3]);
    a = an; b0 = bn0; b1 = bn1; b2 = bn2; b3 = bn3;
  }
#pragma unroll
  for (int j = 0; j < 4; ++j) {
    int n = nb * 64 + j * 16 + (lane & 15);
    int h = n >> 6, d = n & 63;
    float bval = bias[n];
#pragma unroll
    for (int r = 0; r < 8; ++r) {
      int m = mb * 16 + r + ((lane >> 4) << 3);
      unsigned short bf = f2bf(c[j][r] + bval);
      if (p == 0)      q_bf[((size_t)h * SEQ + m) * HD + d] = bf;
      else if (p == 1) k_bf[((size_t)h * SEQ + m) * HD + d] = bf;
      else             vt_bf[((size_t)h * HD + d) * SEQ + m] = bf;
    }
  }
}

// ============ kernel 2: scores = QK^T/sqrt(D), causal mask, 32x64 tile/wave ====
// All 12 A/B fragments loaded up front (K=64 -> 2 k-steps), then 16 WMMAs
// back-to-back: maximum matrix-pipe density for the dominant GEMM.
__global__ __launch_bounds__(256) void mha_scores_kernel(
    const unsigned short* __restrict__ q_bf,
    const unsigned short* __restrict__ k_bf,
    float* __restrict__ attn) {
  const float NEG_INF = -__builtin_inff();
  int lane = threadIdx.x & 31;
  int wid  = (blockIdx.x * blockDim.x + threadIdx.x) >> 5;
  int nb = wid & 63;                           // 64 n-blocks of 64
  int mb = (wid >> 6) & 127;                   // 128 m-blocks of 32
  int h  = wid >> 13;                          // 8 heads
  int m0 = mb * 32, n0 = nb * 64;
  float* out = attn + (size_t)h * SEQ * SEQ;

  if (n0 > m0 + 31) {                          // tile fully above the diagonal
#pragma unroll
    for (int j = 0; j < 4; ++j) {
      int n = n0 + j * 16 + (lane & 15);
#pragma unroll
      for (int r = 0; r < 8; ++r) {
        int mr = r + ((lane >> 4) << 3);
        out[(size_t)(m0 + mr) * SEQ + n]      = NEG_INF;
        out[(size_t)(m0 + 16 + mr) * SEQ + n] = NEG_INF;
      }
    }
    return;
  }

  const unsigned short* A0 = q_bf + ((size_t)h * SEQ + m0) * HD;
  const unsigned short* A1 = A0 + 16 * HD;
  const unsigned short* BT = k_bf + (size_t)h * SEQ * HD;   // (S,D) == N-major

  // issue every fragment load before any WMMA
  v16bf a00 = load_a_bf16(A0, HD, 0, lane);
  v16bf a01 = load_a_bf16(A0, HD, 32, lane);
  v16bf a10 = load_a_bf16(A1, HD, 0, lane);
  v16bf a11 = load_a_bf16(A1, HD, 32, lane);
  v16bf bk0[4], bk1[4];
#pragma unroll
  for (int j = 0; j < 4; ++j) {
    const unsigned short* Bj = BT + (size_t)(n0 + j * 16) * HD;
    bk0[j] = load_b_bf16(Bj, HD, 0, lane);
    bk1[j] = load_b_bf16(Bj, HD, 32, lane);
  }

  v8f c[8] = {};
#pragma unroll
  for (int j = 0; j < 4; ++j) {
    c[j]     = wmma_bf16(a00, bk0[j], c[j]);
    c[j]     = wmma_bf16(a01, bk1[j], c[j]);
    c[4 + j] = wmma_bf16(a10, bk0[j], c[4 + j]);
    c[4 + j] = wmma_bf16(a11, bk1[j], c[4 + j]);
  }

  const float scale = 0.125f;                  // 1/sqrt(64)
#pragma unroll
  for (int half = 0; half < 2; ++half) {
#pragma unroll
    for (int j = 0; j < 4; ++j) {
      int n = n0 + j * 16 + (lane & 15);
#pragma unroll
      for (int r = 0; r < 8; ++r) {
        int m = m0 + half * 16 + r + ((lane >> 4) << 3);
        out[(size_t)m * SEQ + n] =
            (n <= m) ? c[half * 4 + j][r] * scale : NEG_INF;
      }
    }
  }
}

// ============ kernel 3: in-place row softmax, one wave per row ============
__global__ __launch_bounds__(256) void mha_softmax_kernel(float* __restrict__ attn) {
  const float NEG_INF = -__builtin_inff();
  int lane = threadIdx.x & 31;
  int row  = (blockIdx.x * blockDim.x + threadIdx.x) >> 5;  // 0 .. NH*SEQ-1
  float* p = attn + (size_t)row * SEQ;

  float mx = NEG_INF, sum = 0.f;
  for (int i = 0; i < SEQ / 128; ++i) {        // 32 lanes * float4
    v4f v = *(const v4f*)(p + (i * 32 + lane) * 4);
#pragma unroll
    for (int j = 0; j < 4; ++j) {
      float x = v[j];
      if (x != NEG_INF) {
        float mn = fmaxf(mx, x);
        sum = sum * __expf(mx - mn) + __expf(x - mn);
        mx = mn;
      }
    }
  }
#pragma unroll
  for (int off = 16; off >= 1; off >>= 1) {    // wave32 reduction
    float mo = __shfl_xor(mx, off, 32);
    float so = __shfl_xor(sum, off, 32);
    if (mo != NEG_INF) {
      float mn = fmaxf(mx, mo);
      sum = sum * __expf(mx - mn) + so * __expf(mo - mn);
      mx = mn;
    }
  }
  float inv = 1.0f / sum;
  for (int i = 0; i < SEQ / 128; ++i) {
    float* q = p + (i * 32 + lane) * 4;
    v4f v = *(const v4f*)q;
#pragma unroll
    for (int j = 0; j < 4; ++j) v[j] = __expf(v[j] - mx) * inv;  // -inf -> 0
    *(v4f*)q = v;
  }
}

// ============ kernel 4: attn @ V (causal-truncated K loop, SW-pipelined) ======
__global__ __launch_bounds__(256) void mha_av_kernel(
    const float* __restrict__ attn,
    const unsigned short* __restrict__ vt_bf,
    unsigned short* __restrict__ ao_bf) {
  int lane = threadIdx.x & 31;
  int wid  = (blockIdx.x * blockDim.x + threadIdx.x) >> 5;
  int mb = wid & 255;
  int h  = wid >> 8;
  int m0 = mb * 16;

  const float*          A  = attn  + ((size_t)h * SEQ + m0) * SEQ;
  const unsigned short* BT = vt_bf + (size_t)h * HD * SEQ;  // (D,S) == N-major
  int kend = m0 + 16;                          // columns > m are exactly 0

  v8f c[4] = {};
  v16bf a  = load_a_f32(A, SEQ, 0, lane);
  v16bf b0 = load_b_bf16(BT,            SEQ, 0, lane);
  v16bf b1 = load_b_bf16(BT + 16 * SEQ, SEQ, 0, lane);
  v16bf b2 = load_b_bf16(BT + 32 * SEQ, SEQ, 0, lane);
  v16bf b3 = load_b_bf16(BT + 48 * SEQ, SEQ, 0, lane);
  for (int kb = 0; kb < kend; kb += 32) {
    int kn = (kb + 32 < kend) ? kb + 32 : kb;
    // the attn stream is the only operand that must come from HBM:
    // prefetch the k-step after next into the cache hierarchy
    __builtin_prefetch((const void*)(A + (size_t)(lane & 15) * SEQ + kn + 32), 0, 1);
    v16bf an  = load_a_f32(A, SEQ, kn, lane);
    v16bf bn0 = load_b_bf16(BT,            SEQ, kn, lane);
    v16bf bn1 = load_b_bf16(BT + 16 * SEQ, SEQ, kn, lane);
    v16bf bn2 = load_b_bf16(BT + 32 * SEQ, SEQ, kn, lane);
    v16bf bn3 = load_b_bf16(BT + 48 * SEQ, SEQ, kn, lane);
    c[0] = wmma_bf16(a, b0, c[0]);
    c[1] = wmma_bf16(a, b1, c[1]);
    c[2] = wmma_bf16(a, b2, c[2]);
    c[3] = wmma_bf16(a, b3, c[3]);
    a = an; b0 = bn0; b1 = bn1; b2 = bn2; b3 = bn3;
  }
#pragma unroll
  for (int j = 0; j < 4; ++j) {
    int d = j * 16 + (lane & 15);
#pragma unroll
    for (int r = 0; r < 8; ++r) {
      int m = m0 + r + ((lane >> 4) << 3);
      ao_bf[(size_t)m * EMB + h * HD + d] = f2bf(c[j][r]);
    }
  }
}

// ============ kernel 5: output projection + bias (fp32 out, SW-pipelined) =====
__global__ __launch_bounds__(256) void mha_oproj_kernel(
    const unsigned short* __restrict__ ao_bf,
    const unsigned short* __restrict__ wot,
    const float* __restrict__ bo,
    float* __restrict__ out) {
  int lane = threadIdx.x & 31;
  int wid  = (blockIdx.x * blockDim.x + threadIdx.x) >> 5;
  int nb = wid & 7;
  int mb = wid >> 3;

  const unsigned short* A  = ao_bf + (size_t)mb * 16 * EMB;
  const unsigned short* B0 = wot + (size_t)(nb * 64) * EMB;

  v8f c[4] = {};
  v16bf a  = load_a_bf16(A, EMB, 0, lane);
  v16bf b0 = load_b_bf16(B0,            EMB, 0, lane);
  v16bf b1 = load_b_bf16(B0 + 16 * EMB, EMB, 0, lane);
  v16bf b2 = load_b_bf16(B0 + 32 * EMB, EMB, 0, lane);
  v16bf b3 = load_b_bf16(B0 + 48 * EMB, EMB, 0, lane);
  for (int kb = 0; kb < EMB; kb += 32) {
    int kn = (kb + 32 < EMB) ? kb + 32 : kb;
    v16bf an  = load_a_bf16(A, EMB, kn, lane);
    v16bf bn0 = load_b_bf16(B0,            EMB, kn, lane);
    v16bf bn1 = load_b_bf16(B0 + 16 * EMB, EMB, kn, lane);
    v16bf bn2 = load_b_bf16(B0 + 32 * EMB, EMB, kn, lane);
    v16bf bn3 = load_b_bf16(B0 + 48 * EMB, EMB, kn, lane);
    c[0] = wmma_bf16(a, b0, c[0]);
    c[1] = wmma_bf16(a, b1, c[1]);
    c[2] = wmma_bf16(a, b2, c[2]);
    c[3] = wmma_bf16(a, b3, c[3]);
    a = an; b0 = bn0; b1 = bn1; b2 = bn2; b3 = bn3;
  }
#pragma unroll
  for (int j = 0; j < 4; ++j) {
    int n = nb * 64 + j * 16 + (lane & 15);
    float bval = bo[n];
#pragma unroll
    for (int r = 0; r < 8; ++r) {
      int m = mb * 16 + r + ((lane >> 4) << 3);
      out[(size_t)m * EMB + n] = c[j][r] + bval;
    }
  }
}

extern "C" void kernel_launch(void* const* d_in, const int* in_sizes, int n_in,
                              void* d_out, int out_size, void* d_ws, size_t ws_size,
                              hipStream_t stream) {
  // inputs per setup_inputs(): x, mask, Wq, bq, Wk, bk, Wv, bv, Wo, bo
  const float* x  = (const float*)d_in[0];
  const float* Wq = (const float*)d_in[2];
  const float* bq = (const float*)d_in[3];
  const float* Wk = (const float*)d_in[4];
  const float* bk = (const float*)d_in[5];
  const float* Wv = (const float*)d_in[6];
  const float* bv = (const float*)d_in[7];
  const float* Wo = (const float*)d_in[8];
  const float* bo = (const float*)d_in[9];
  (void)in_sizes; (void)n_in; (void)out_size;

  float* out  = (float*)d_out;                 // (1,4096,512)
  float* attn = out + (size_t)SEQ * EMB;       // (1,8,4096,4096)

  // workspace layout (bf16 stored as u16); total ~22 MB
  unsigned short* ws   = (unsigned short*)d_ws;
  unsigned short* x_bf = ws;                               // 4096*512
  unsigned short* wqt  = x_bf + (size_t)SEQ * EMB;         // 512*512
  unsigned short* wkt  = wqt + (size_t)EMB * EMB;
  unsigned short* wvt  = wkt + (size_t)EMB * EMB;
  unsigned short* wot  = wvt + (size_t)EMB * EMB;
  unsigned short* q_bf = wot + (size_t)EMB * EMB;          // (H,S,D)
  unsigned short* k_bf = q_bf + (size_t)NH * SEQ * HD;     // (H,S,D)
  unsigned short* vt_bf= k_bf + (size_t)NH * SEQ * HD;     // (H,D,S)
  unsigned short* ao_bf= vt_bf + (size_t)NH * HD * SEQ;    // (S,E)
  (void)ws_size;

  // 0) conversions / transposes
  mha_prep_kernel<<<(SEQ * EMB + 255) / 256, 256, 0, stream>>>(
      x, Wq, Wk, Wv, Wo, x_bf, wqt, wkt, wvt, wot);

  // 1) QKV projections: 3 * 256 * 8 = 6144 waves
  mha_qkv_kernel<<<6144 * 32 / 256, 256, 0, stream>>>(
      x_bf, wqt, wkt, wvt, bq, bk, bv, q_bf, k_bf, vt_bf);

  // 2) scores: 8 heads * 128 m-blocks(32) * 64 n-blocks(64) = 65536 waves
  mha_scores_kernel<<<65536 * 32 / 256, 256, 0, stream>>>(q_bf, k_bf, attn);

  // 3) softmax: one wave per row, 8*4096 rows
  mha_softmax_kernel<<<NH * SEQ * 32 / 256, 256, 0, stream>>>(attn);

  // 4) attn @ V: 8 * 256 = 2048 waves
  mha_av_kernel<<<2048 * 32 / 256, 256, 0, stream>>>(attn, vt_bf, ao_bf);

  // 5) output projection: 256 * 8 = 2048 waves
  mha_oproj_kernel<<<2048 * 32 / 256, 256, 0, stream>>>(ao_bf, wot, bo, out);
}
